// NERModel_30081950941376
// MI455X (gfx1250) — compile-verified, matrix operand order checked by reading, not run on hardware
//
#include <hip/hip_runtime.h>
#include <math.h>

// ---------------------------------------------------------------------------
// BERT-NER + CRF for MI455X (gfx1250), wave32 + WMMA bf16 + async-LDS staging.
// B=16 S=512 H=768 NH=12 HD=64 F=3072 NL=12 L=9
// ---------------------------------------------------------------------------

typedef __bf16 bf16;
typedef __attribute__((ext_vector_type(16))) __bf16 v16bf;
typedef __attribute__((ext_vector_type(8)))  __bf16 v8bf;
typedef __attribute__((ext_vector_type(8)))  float  v8f;

#define Bsz 16
#define Ssz 512
#define Hsz 768
#define NHsz 12
#define HDsz 64
#define Fsz 3072
#define NLsz 12
#define Lsz 9
#define NTOK (Bsz * Ssz)        // 8192
#define Tcrf (Ssz - 1)          // 511

// Async global->LDS staging (gfx1250 GLOBAL_LOAD_ASYNC_TO_LDS_*, ASYNCcnt).
#if defined(__has_builtin)
#if __has_builtin(__builtin_amdgcn_global_load_async_to_lds_b128) && \
    __has_builtin(__builtin_amdgcn_s_wait_asynccnt)
#define USE_ASYNC_LDS 1
#endif
#endif

#define AS1 __attribute__((address_space(1)))
#define AS3 __attribute__((address_space(3)))

#if defined(USE_ASYNC_LDS)
// Builtin signature (from clang diagnostic): (v4i AS1*, v4i AS3*, Imm, Imm)
typedef __attribute__((ext_vector_type(4))) int i4_t;
typedef AS1 i4_t* gptr_b128;
typedef AS3 i4_t* lptr_b128;
__device__ __forceinline__ void async_copy_b128(const bf16* g, bf16* l) {
    __builtin_amdgcn_global_load_async_to_lds_b128((gptr_b128)(g), (lptr_b128)(l), 0, 0);
}
#endif

// ISA 7.12.2, 16-bit A/B fragment: lane (x = lane&15, h = lane>>4) holds
// K = {8h..8h+7} then {16+8h..16+8h+7} -- two contiguous 16-byte runs when
// the tile row (32 K-values) is stored contiguously. Load as 2x b128.
__device__ __forceinline__ v16bf load_frag(const bf16* __restrict__ row32, int h) {
    v8bf lo = *(const v8bf*)(row32 + 8 * h);
    v8bf hi = *(const v8bf*)(row32 + 16 + 8 * h);
    return __builtin_shufflevector(lo, hi, 0, 1, 2, 3, 4, 5, 6, 7,
                                   8, 9, 10, 11, 12, 13, 14, 15);
}

__device__ __forceinline__ v8f wmma_bf16(v16bf a, v16bf b, v8f c) {
    // (neg_a, A, neg_b, B, c_mod, C, reuse_a, reuse_b)
    return __builtin_amdgcn_wmma_f32_16x16x32_bf16(false, a, false, b, (short)0, c,
                                                   false, false);
}

// ---------------------------------------------------------------------------
// fp32 -> bf16 conversion
// ---------------------------------------------------------------------------
__global__ __launch_bounds__(256) void f32_to_bf16_kernel(const float* __restrict__ in,
                                                          bf16* __restrict__ out,
                                                          int n) {
    int i = blockIdx.x * 256 + threadIdx.x;
    if (i < n) out[i] = (bf16)in[i];
}

// ---------------------------------------------------------------------------
// Generic WMMA GEMM: C[M,N] = A[M,K](bf16) @ B[K,N](bf16) + bias, opt. GELU.
// Block tile 128x64, 8 waves (256 thr), each wave does 16(M) x 64(N).
// A tile: async DMA to LDS row-major (m x 32K).  B tile: staged TRANSPOSED
// (n x 32K) via VGPRs.  Every fragment is two ds_load_b128.
// ---------------------------------------------------------------------------
__global__ __launch_bounds__(256) void gemm_bf16_kernel(const bf16* __restrict__ A,
                                                        const bf16* __restrict__ Bm,
                                                        const float* __restrict__ bias,
                                                        float* __restrict__ Cf,
                                                        bf16* __restrict__ Cb,
                                                        int M, int N, int K, int act) {
    __shared__ __align__(16) bf16 sA[128 * 32];   // [m][k]
    __shared__ __align__(16) bf16 sBt[64 * 32];   // [n][k] (transposed)
    const int tid  = threadIdx.x;
    const int lane = tid & 31;
    const int wave = tid >> 5;
    const int h    = lane >> 4;
    const int nlo  = lane & 15;
    const int m0 = blockIdx.x * 128;
    const int n0 = blockIdx.y * 64;

    v8f acc0 = {}, acc1 = {}, acc2 = {}, acc3 = {};

    for (int k0 = 0; k0 < K; k0 += 32) {
        // ---- Stage A tile 128x32: 512 x b128 chunks (8 bf16 each) ---------
#if defined(USE_ASYNC_LDS)
#pragma unroll
        for (int j = 0; j < 2; ++j) {
            int c = tid + j * 256;
            int row = c >> 2, col = (c & 3) << 3;
            async_copy_b128(A + (size_t)(m0 + row) * K + k0 + col, sA + c * 8);
        }
#else
        {
            int c0 = tid, c1 = tid + 256;
            uint4 t0 = *(const uint4*)(A + (size_t)(m0 + (c0 >> 2)) * K + k0 + ((c0 & 3) << 3));
            uint4 t1 = *(const uint4*)(A + (size_t)(m0 + (c1 >> 2)) * K + k0 + ((c1 & 3) << 3));
            ((uint4*)sA)[c0] = t0;
            ((uint4*)sA)[c1] = t1;
        }
#endif
        // ---- Stage B tile 32x64 transposed into sBt[n][k] -----------------
        {
            int kk = tid >> 3, n8 = (tid & 7) << 3;
            union { uint4 u; bf16 e[8]; } d;
            d.u = *(const uint4*)(Bm + (size_t)(k0 + kk) * N + n0 + n8);
#pragma unroll
            for (int i = 0; i < 8; ++i) sBt[(n8 + i) * 32 + kk] = d.e[i];
        }
        // Prefetch next K-step tiles into cache (global_prefetch_b8)
        if (k0 + 32 < K) {
            __builtin_prefetch(A + (size_t)(m0 + (tid >> 1)) * K + k0 + 32, 0, 1);
            __builtin_prefetch(Bm + (size_t)(k0 + 32 + (tid & 31)) * N + n0, 0, 1);
        }
#if defined(USE_ASYNC_LDS)
        __builtin_amdgcn_s_wait_asynccnt(0);
#endif
        __syncthreads();

        const bf16* arow = sA + ((wave << 4) + nlo) * 32;
        v16bf af = load_frag(arow, h);
        v16bf b0 = load_frag(sBt + (nlo + 0)  * 32, h);
        v16bf b1 = load_frag(sBt + (nlo + 16) * 32, h);
        v16bf b2 = load_frag(sBt + (nlo + 32) * 32, h);
        v16bf b3 = load_frag(sBt + (nlo + 48) * 32, h);
        acc0 = wmma_bf16(af, b0, acc0);
        acc1 = wmma_bf16(af, b1, acc1);
        acc2 = wmma_bf16(af, b2, acc2);
        acc3 = wmma_bf16(af, b3, acc3);
        __syncthreads();
    }

    // Epilogue. D layout: lane -> (M = v + 8*(lane>>4), N = lane&15)
    const int mbase = m0 + (wave << 4) + (h << 3);
    const int nb    = n0 + nlo;
    v8f accs[4] = {acc0, acc1, acc2, acc3};
#pragma unroll
    for (int nt = 0; nt < 4; ++nt) {
        int col = nb + nt * 16;
        float bv = bias ? bias[col] : 0.0f;
#pragma unroll
        for (int v = 0; v < 8; ++v) {
            float x = accs[nt][v] + bv;
            if (act == 1) x = 0.5f * x * (1.0f + erff(x * 0.70710678118f));
            size_t idx = (size_t)(mbase + v) * N + col;
            if (Cf) Cf[idx] = x;
            if (Cb) Cb[idx] = (bf16)x;
        }
    }
}

// ---------------------------------------------------------------------------
// Fused attention: one block = (16 query rows, one head, one batch).
// Phase1: S = Q K^T / 8 via WMMA -> LDS; Phase2: softmax; Phase3: P V via WMMA
// with V slabs staged transposed in LDS. Q/K fragments are 2x global b128.
// qkv layout: [B, S, 3H], q at h*64, k at H + h*64, v at 2H + h*64.
// ---------------------------------------------------------------------------
__global__ __launch_bounds__(128) void attention_kernel(const bf16* __restrict__ qkv,
                                                        bf16* __restrict__ ctx) {
    __shared__ __align__(16) float sS[16 * Ssz];
    __shared__ __align__(16) bf16  sP[16 * Ssz];
    __shared__ __align__(16) bf16  sV[64 * 32];   // [d][kk] per 32-step slab
    __shared__ float sRed[16][8];
    __shared__ float sMax[16];
    __shared__ float sSum[16];

    const int qt = blockIdx.x;      // query tile (0..31)
    const int hh = blockIdx.y;      // head
    const int b  = blockIdx.z;      // batch
    const int tid  = threadIdx.x;
    const int lane = tid & 31;
    const int wave = tid >> 5;      // 0..3
    const int hl   = lane >> 4;
    const int nlo  = lane & 15;
    const size_t rs   = 3 * Hsz;    // qkv row stride
    const size_t base = (size_t)b * Ssz * rs;

    // ---- Phase 1: scores --------------------------------------------------
    const bf16* qp = qkv + base + (size_t)(qt * 16 + nlo) * rs + hh * HDsz;
    v16bf aq0 = load_frag(qp, hl);
    v16bf aq1 = load_frag(qp + 32, hl);
    for (int nt = wave * 8; nt < wave * 8 + 8; ++nt) {
        const bf16* kp = qkv + base + (size_t)(nt * 16 + nlo) * rs + Hsz + hh * HDsz;
        v8f acc = {};
        acc = wmma_bf16(aq0, load_frag(kp, hl), acc);
        acc = wmma_bf16(aq1, load_frag(kp + 32, hl), acc);
        const int mb = hl << 3;
#pragma unroll
        for (int v = 0; v < 8; ++v)
            sS[(mb + v) * Ssz + nt * 16 + nlo] = acc[v] * 0.125f; // 1/sqrt(64)
    }
    __syncthreads();

    // ---- Phase 2: softmax (16 rows x 512, 8 threads/row) ------------------
    {
        const int r = tid >> 3, sub = tid & 7;
        float mx = -1e30f;
        for (int c = sub; c < Ssz; c += 8) mx = fmaxf(mx, sS[r * Ssz + c]);
        sRed[r][sub] = mx;
        __syncthreads();
        if (sub == 0) {
            float m2 = sRed[r][0];
            for (int i = 1; i < 8; ++i) m2 = fmaxf(m2, sRed[r][i]);
            sMax[r] = m2;
        }
        __syncthreads();
        const float m2 = sMax[r];
        float sum = 0.0f;
        for (int c = sub; c < Ssz; c += 8) {
            float e = __expf(sS[r * Ssz + c] - m2);
            sS[r * Ssz + c] = e;
            sum += e;
        }
        sRed[r][sub] = sum;
        __syncthreads();
        if (sub == 0) {
            float s2 = 0.0f;
            for (int i = 0; i < 8; ++i) s2 += sRed[r][i];
            sSum[r] = 1.0f / s2;
        }
        __syncthreads();
        const float inv = sSum[r];
        for (int c = sub; c < Ssz; c += 8)
            sP[r * Ssz + c] = (bf16)(sS[r * Ssz + c] * inv);
    }

    // ---- Phase 3: ctx = P @ V; wave w covers d-cols [w*16, w*16+16) -------
    {
        v8f acc = {};
        for (int ks = 0; ks < 16; ++ks) {
            // Stage V slab (rows ks*32..+31, 64 cols) transposed: sV[d][kk]
            __syncthreads();
            {
                int c0 = tid, c1 = tid + 128;
                union { uint4 u; bf16 e[8]; } d0, d1;
                int kk0 = c0 >> 3, d80 = (c0 & 7) << 3;
                int kk1 = c1 >> 3, d81 = (c1 & 7) << 3;
                d0.u = *(const uint4*)(qkv + base + (size_t)(ks * 32 + kk0) * rs +
                                       2 * Hsz + hh * HDsz + d80);
                d1.u = *(const uint4*)(qkv + base + (size_t)(ks * 32 + kk1) * rs +
                                       2 * Hsz + hh * HDsz + d81);
#pragma unroll
                for (int i = 0; i < 8; ++i) sV[(d80 + i) * 32 + kk0] = d0.e[i];
#pragma unroll
                for (int i = 0; i < 8; ++i) sV[(d81 + i) * 32 + kk1] = d1.e[i];
            }
            __syncthreads();
            v16bf ap = load_frag(sP + nlo * Ssz + ks * 32, hl);
            v16bf bv = load_frag(sV + (wave * 16 + nlo) * 32, hl);
            acc = wmma_bf16(ap, bv, acc);
        }
        const int mb = hl << 3;
#pragma unroll
        for (int v = 0; v < 8; ++v) {
            int q = qt * 16 + mb + v;
            ctx[((size_t)b * Ssz + q) * Hsz + hh * HDsz + wave * 16 + nlo] = (bf16)acc[v];
        }
    }
}

// ---------------------------------------------------------------------------
// Embedding + LayerNorm (one block per token row, 256 thr, 3 elems/thr)
// ---------------------------------------------------------------------------
__global__ __launch_bounds__(256) void embed_ln_kernel(const int* __restrict__ ids,
                                                       const float* __restrict__ we,
                                                       const float* __restrict__ pe,
                                                       const float* __restrict__ te,
                                                       const float* __restrict__ g,
                                                       const float* __restrict__ beta,
                                                       float* __restrict__ x,
                                                       bf16* __restrict__ xb) {
    const int row = blockIdx.x;
    const int s   = row % Ssz;
    const int tid = threadIdx.x;
    const int id  = ids[row];
    __shared__ float red[256];

    float v[3];
    float lsum = 0.0f;
#pragma unroll
    for (int j = 0; j < 3; ++j) {
        int i = tid + j * 256;
        float val = we[(size_t)id * Hsz + i] + pe[(size_t)s * Hsz + i] + te[i];
        v[j] = val;
        lsum += val;
    }
    red[tid] = lsum; __syncthreads();
    for (int o = 128; o > 0; o >>= 1) { if (tid < o) red[tid] += red[tid + o]; __syncthreads(); }
    const float mu = red[0] * (1.0f / Hsz);
    __syncthreads();
    float lvar = 0.0f;
#pragma unroll
    for (int j = 0; j < 3; ++j) { float d = v[j] - mu; lvar += d * d; }
    red[tid] = lvar; __syncthreads();
    for (int o = 128; o > 0; o >>= 1) { if (tid < o) red[tid] += red[tid + o]; __syncthreads(); }
    const float rstd = rsqrtf(red[0] * (1.0f / Hsz) + 1e-12f);
#pragma unroll
    for (int j = 0; j < 3; ++j) {
        int i = tid + j * 256;
        float y = (v[j] - mu) * rstd * g[i] + beta[i];
        x[(size_t)row * Hsz + i]  = y;
        xb[(size_t)row * Hsz + i] = (bf16)y;
    }
}

// ---------------------------------------------------------------------------
// Residual add + LayerNorm: out = LN(a + d) * g + beta   (in-place safe on a)
// ---------------------------------------------------------------------------
__global__ __launch_bounds__(256) void residual_ln_kernel(const float* __restrict__ a,
                                                          const float* __restrict__ d,
                                                          const float* __restrict__ g,
                                                          const float* __restrict__ beta,
                                                          float* __restrict__ out,
                                                          bf16* __restrict__ outb) {
    const int row = blockIdx.x;
    const int tid = threadIdx.x;
    __shared__ float red[256];

    float v[3];
    float lsum = 0.0f;
#pragma unroll
    for (int j = 0; j < 3; ++j) {
        int i = tid + j * 256;
        float val = a[(size_t)row * Hsz + i] + d[(size_t)row * Hsz + i];
        v[j] = val;
        lsum += val;
    }
    red[tid] = lsum; __syncthreads();
    for (int o = 128; o > 0; o >>= 1) { if (tid < o) red[tid] += red[tid + o]; __syncthreads(); }
    const float mu = red[0] * (1.0f / Hsz);
    __syncthreads();
    float lvar = 0.0f;
#pragma unroll
    for (int j = 0; j < 3; ++j) { float dd = v[j] - mu; lvar += dd * dd; }
    red[tid] = lvar; __syncthreads();
    for (int o = 128; o > 0; o >>= 1) { if (tid < o) red[tid] += red[tid + o]; __syncthreads(); }
    const float rstd = rsqrtf(red[0] * (1.0f / Hsz) + 1e-12f);
#pragma unroll
    for (int j = 0; j < 3; ++j) {
        int i = tid + j * 256;
        float y = (v[j] - mu) * rstd * g[i] + beta[i];
        out[(size_t)row * Hsz + i]  = y;
        outb[(size_t)row * Hsz + i] = (bf16)y;
    }
}

// ---------------------------------------------------------------------------
// Classifier: logits[token, l] = x . cls_w[:, l] + cls_b[l]  (L=9, tiny)
// ---------------------------------------------------------------------------
__global__ __launch_bounds__(256) void classifier_kernel(const float* __restrict__ x,
                                                         const float* __restrict__ w,
                                                         const float* __restrict__ bias,
                                                         float* __restrict__ logits) {
    int idx = blockIdx.x * 256 + threadIdx.x;
    if (idx >= NTOK * Lsz) return;
    int row = idx / Lsz, l = idx % Lsz;
    float s = bias[l];
    const float* xr = x + (size_t)row * Hsz;
    for (int i = 0; i < Hsz; ++i) s += xr[i] * w[i * Lsz + l];
    logits[idx] = s;
}

// ---------------------------------------------------------------------------
// Zero loss accumulator
// ---------------------------------------------------------------------------
__global__ void zero_loss_kernel(float* p) { if (threadIdx.x == 0) *p = 0.0f; }

// ---------------------------------------------------------------------------
// CRF: forward logsumexp + Viterbi + backtrace. One block (1 wave) per batch.
// em[b][t][l] = logits[b][t+1][l]; mask is all-true (ids >= 1).
// ---------------------------------------------------------------------------
__global__ __launch_bounds__(32) void crf_kernel(const float* __restrict__ logits,
                                                 const int* __restrict__ labels,
                                                 const float* __restrict__ start,
                                                 const float* __restrict__ end,
                                                 const float* __restrict__ trans,
                                                 float* __restrict__ pred,
                                                 float* __restrict__ tagsf,
                                                 float* __restrict__ loss,
                                                 int* __restrict__ bp) {
    const int b = blockIdx.x;
    const int l = threadIdx.x;
    __shared__ float alpha[Lsz], score[Lsz], na[Lsz], ns[Lsz], tr[Lsz * Lsz];

    for (int i = l; i < Lsz * Lsz; i += 32) tr[i] = trans[i];
    __syncthreads();

    const float* em = logits + ((size_t)b * Ssz + 1) * Lsz; // em(t,l)=em[t*L+l]
    if (l < Lsz) {
        alpha[l] = start[l] + em[l];
        score[l] = alpha[l];
    }
    __syncthreads();

    int* bpb = bp + (size_t)b * Tcrf * Lsz;
    for (int t = 1; t < Tcrf; ++t) {
        if (l < Lsz) {
            float e = em[t * Lsz + l];
            float mx = -1e30f;
            for (int i = 0; i < Lsz; ++i) mx = fmaxf(mx, alpha[i] + tr[i * Lsz + l]);
            float s = 0.0f;
            for (int i = 0; i < Lsz; ++i) s += __expf(alpha[i] + tr[i * Lsz + l] - mx);
            na[l] = mx + __logf(s) + e;
            float bs = -1e30f; int bi = 0;
            for (int i = 0; i < Lsz; ++i) {
                float c = score[i] + tr[i * Lsz + l];
                if (c > bs) { bs = c; bi = i; }
            }
            ns[l] = bs + e;
            bpb[t * Lsz + l] = bi;
        }
        __syncthreads();
        if (l < Lsz) { alpha[l] = na[l]; score[l] = ns[l]; }
        __syncthreads();
    }

    if (l == 0) {
        // logZ
        float mx = -1e30f;
        for (int j = 0; j < Lsz; ++j) mx = fmaxf(mx, alpha[j] + end[j]);
        float s = 0.0f;
        for (int j = 0; j < Lsz; ++j) s += __expf(alpha[j] + end[j] - mx);
        float logZ = mx + __logf(s);
        // numerator along gold path
        const int* lb = labels + (size_t)b * Ssz + 1;
        float num = start[lb[0]] + em[lb[0]];
        for (int t = 1; t < Tcrf; ++t)
            num += tr[lb[t - 1] * Lsz + lb[t]] + em[t * Lsz + lb[t]];
        num += end[lb[Tcrf - 1]];
        atomicAdd(loss, logZ - num);  // loss = -sum(llh)
        // Viterbi backtrace
        float bs = -1e30f; int cur = 0;
        for (int j = 0; j < Lsz; ++j) {
            float c = score[j] + end[j];
            if (c > bs) { bs = c; cur = j; }
        }
        pred[(size_t)b * Tcrf + Tcrf - 1] = (float)cur;
        for (int t = Tcrf - 1; t >= 1; --t) {
            cur = bpb[t * Lsz + cur];
            pred[(size_t)b * Tcrf + t - 1] = (float)cur;
        }
    }
    for (int t = l; t < Tcrf; t += 32)
        tagsf[(size_t)b * Tcrf + t] = (float)labels[(size_t)b * Ssz + 1 + t];
}

// ---------------------------------------------------------------------------
// Host orchestration
// ---------------------------------------------------------------------------
extern "C" void kernel_launch(void* const* d_in, const int* in_sizes, int n_in,
                              void* d_out, int out_size, void* d_ws, size_t ws_size,
                              hipStream_t stream) {
    const int*   input_ids = (const int*)d_in[0];
    const int*   labels    = (const int*)d_in[1];
    const float* word_emb  = (const float*)d_in[2];
    const float* pos_emb   = (const float*)d_in[3];
    const float* type_emb  = (const float*)d_in[4];
    const float* emb_ln_g  = (const float*)d_in[5];
    const float* emb_ln_b  = (const float*)d_in[6];
    const float* qkv_w     = (const float*)d_in[7];
    const float* qkv_b     = (const float*)d_in[8];
    const float* ao_w      = (const float*)d_in[9];
    const float* ao_b      = (const float*)d_in[10];
    const float* ln1_g     = (const float*)d_in[11];
    const float* ln1_b     = (const float*)d_in[12];
    const float* f1_w      = (const float*)d_in[13];
    const float* f1_b      = (const float*)d_in[14];
    const float* f2_w      = (const float*)d_in[15];
    const float* f2_b      = (const float*)d_in[16];
    const float* ln2_g     = (const float*)d_in[17];
    const float* ln2_b     = (const float*)d_in[18];
    const float* cls_w     = (const float*)d_in[19];
    const float* cls_b     = (const float*)d_in[20];
    const float* crf_start = (const float*)d_in[21];
    const float* crf_end   = (const float*)d_in[22];
    const float* crf_trans = (const float*)d_in[23];

    // Workspace carve-out
    char* w = (char*)d_ws;
    size_t off = 0;
    auto carve = [&](size_t bytes) -> void* {
        void* p = w + off;
        off = (off + bytes + 255) & ~(size_t)255;
        return p;
    };
    float* x      = (float*)carve((size_t)NTOK * Hsz * 4);
    bf16*  xb     = (bf16*) carve((size_t)NTOK * Hsz * 2);
    bf16*  qkvb   = (bf16*) carve((size_t)NTOK * 3 * Hsz * 2);
    bf16*  ctxb   = (bf16*) carve((size_t)NTOK * Hsz * 2);
    bf16*  ffhb   = (bf16*) carve((size_t)NTOK * Fsz * 2);
    float* tmp    = (float*)carve((size_t)NTOK * Hsz * 4);
    bf16*  wqkv   = (bf16*) carve((size_t)Hsz * 3 * Hsz * 2);
    bf16*  wao    = (bf16*) carve((size_t)Hsz * Hsz * 2);
    bf16*  wf1    = (bf16*) carve((size_t)Hsz * Fsz * 2);
    bf16*  wf2    = (bf16*) carve((size_t)Fsz * Hsz * 2);
    float* logits = (float*)carve((size_t)NTOK * Lsz * 4);
    int*   bp     = (int*)  carve((size_t)Bsz * Tcrf * Lsz * 4);

    float* out_pred = (float*)d_out;                       // B*(S-1)
    float* out_tags = out_pred + (size_t)Bsz * Tcrf;       // B*(S-1)
    float* out_loss = out_tags + (size_t)Bsz * Tcrf;       // scalar

    zero_loss_kernel<<<1, 32, 0, stream>>>(out_loss);

    // Embedding + LN
    embed_ln_kernel<<<NTOK, 256, 0, stream>>>(input_ids, word_emb, pos_emb, type_emb,
                                              emb_ln_g, emb_ln_b, x, xb);

    const int nqkvw = Hsz * 3 * Hsz;  // 1,769,472
    const int naow  = Hsz * Hsz;      //   589,824
    const int nf1w  = Hsz * Fsz;      // 2,359,296
    const int nf2w  = Fsz * Hsz;      // 2,359,296

    for (int l = 0; l < NLsz; ++l) {
        // Convert this layer's weights to bf16
        f32_to_bf16_kernel<<<(nqkvw + 255) / 256, 256, 0, stream>>>(qkv_w + (size_t)l * nqkvw, wqkv, nqkvw);
        f32_to_bf16_kernel<<<(naow  + 255) / 256, 256, 0, stream>>>(ao_w  + (size_t)l * naow,  wao,  naow);
        f32_to_bf16_kernel<<<(nf1w  + 255) / 256, 256, 0, stream>>>(f1_w  + (size_t)l * nf1w,  wf1,  nf1w);
        f32_to_bf16_kernel<<<(nf2w  + 255) / 256, 256, 0, stream>>>(f2_w  + (size_t)l * nf2w,  wf2,  nf2w);

        // QKV projection: [8192,768] @ [768,2304] -> bf16 qkv
        gemm_bf16_kernel<<<dim3(NTOK / 128, 3 * Hsz / 64), 256, 0, stream>>>(
            xb, wqkv, qkv_b + (size_t)l * 3 * Hsz, nullptr, qkvb, NTOK, 3 * Hsz, Hsz, 0);

        // Fused attention -> ctx (bf16)
        attention_kernel<<<dim3(Ssz / 16, NHsz, Bsz), 128, 0, stream>>>(qkvb, ctxb);

        // Attention output proj: [8192,768] @ [768,768] -> tmp (f32)
        gemm_bf16_kernel<<<dim3(NTOK / 128, Hsz / 64), 256, 0, stream>>>(
            ctxb, wao, ao_b + (size_t)l * Hsz, tmp, nullptr, NTOK, Hsz, Hsz, 0);

        residual_ln_kernel<<<NTOK, 256, 0, stream>>>(x, tmp, ln1_g + (size_t)l * Hsz,
                                                     ln1_b + (size_t)l * Hsz, x, xb);

        // FFN up + GELU: [8192,768] @ [768,3072] -> bf16
        gemm_bf16_kernel<<<dim3(NTOK / 128, Fsz / 64), 256, 0, stream>>>(
            xb, wf1, f1_b + (size_t)l * Fsz, nullptr, ffhb, NTOK, Fsz, Hsz, 1);

        // FFN down: [8192,3072] @ [3072,768] -> tmp (f32)
        gemm_bf16_kernel<<<dim3(NTOK / 128, Hsz / 64), 256, 0, stream>>>(
            ffhb, wf2, f2_b + (size_t)l * Hsz, tmp, nullptr, NTOK, Hsz, Fsz, 0);

        residual_ln_kernel<<<NTOK, 256, 0, stream>>>(x, tmp, ln2_g + (size_t)l * Hsz,
                                                     ln2_b + (size_t)l * Hsz, x, xb);
    }

    // Classifier + CRF
    classifier_kernel<<<(NTOK * Lsz + 255) / 256, 256, 0, stream>>>(x, cls_w, cls_b, logits);
    crf_kernel<<<Bsz, 32, 0, stream>>>(logits, labels, crf_start, crf_end, crf_trans,
                                       out_pred, out_tags, out_loss, bp);
}
